// GroupQueryAttention_44787918962714
// MI455X (gfx1250) — matrix operand order repaired and checked by reading, not compiled, not run on hardware
//
#include <hip/hip_runtime.h>
#include <stdint.h>

typedef __attribute__((ext_vector_type(16))) __bf16 v16bf;
typedef __attribute__((ext_vector_type(8)))  __bf16 v8bf;
typedef __attribute__((ext_vector_type(4)))  __bf16 v4bf;
typedef __attribute__((ext_vector_type(8)))  float  v8f;
typedef __attribute__((ext_vector_type(4)))  int    v4i_t;

#define HIDDEN 4096
#define NHEAD  32
#define NKV    8
#define HD     128
#define KV_DIM 1024

// ---------------------------------------------------------------------------
// CDNA5 async global->LDS staging (ASYNCcnt path), guarded so the kernel
// still compiles if the toolchain lacks the builtins. Probe round 2 showed
// param0 is `v4i addrspace(1)*` (non-const), so cast accordingly.
// ---------------------------------------------------------------------------
#if defined(__has_builtin)
#if __has_builtin(__builtin_amdgcn_global_load_async_to_lds_b128)
#define HAVE_ASYNC_LDS 1
#endif
#endif
#ifndef HAVE_ASYNC_LDS
#define HAVE_ASYNC_LDS 0
#endif

#if HAVE_ASYNC_LDS && defined(__HIP_DEVICE_COMPILE__)
typedef __attribute__((address_space(1))) v4i_t* gas_v4i_p;
typedef __attribute__((address_space(3))) v4i_t* las_v4i_p;
#endif

__device__ __forceinline__ void copy_b128_to_lds(const __bf16* g, __bf16* l) {
#if HAVE_ASYNC_LDS && defined(__HIP_DEVICE_COMPILE__)
  __builtin_amdgcn_global_load_async_to_lds_b128(
      (gas_v4i_p)(uintptr_t)g,
      (las_v4i_p)(unsigned int)(uintptr_t)l, 0, 0);
#else
  *(v8bf*)l = *(const v8bf*)g;
#endif
}

__device__ __forceinline__ void wait_async_copies() {
#if HAVE_ASYNC_LDS && defined(__HIP_DEVICE_COMPILE__)
#if __has_builtin(__builtin_amdgcn_s_wait_asynccnt)
  __builtin_amdgcn_s_wait_asynccnt(0);
#else
  asm volatile("s_wait_asynccnt 0" ::: "memory");
#endif
#endif
}

// ---------------------------------------------------------------------------
// Elementwise f32 -> bf16 convert (for x / attention activations).
// ---------------------------------------------------------------------------
__global__ __launch_bounds__(256) void convert_f32_bf16(
    const float* __restrict__ s, __bf16* __restrict__ d, int n4) {
  int i = blockIdx.x * blockDim.x + threadIdx.x;
  if (i < n4) {
    float4 v = ((const float4*)s)[i];
    v4bf o;
    o[0] = (__bf16)v.x; o[1] = (__bf16)v.y;
    o[2] = (__bf16)v.z; o[3] = (__bf16)v.w;
    ((v4bf*)d)[i] = o;
  }
}

// ---------------------------------------------------------------------------
// Tiled transpose + convert: src f32 [K][N] -> dst bf16 [N][K].
// Block (32,8), 32x32 tile through LDS; both sides coalesced.
// ---------------------------------------------------------------------------
__global__ __launch_bounds__(256) void transpose_f32_bf16(
    const float* __restrict__ src, __bf16* __restrict__ dst, int K, int N) {
  __shared__ float t[32][33];
  const int bx = blockIdx.x * 32;  // N
  const int by = blockIdx.y * 32;  // K
  const int tx = threadIdx.x, ty = threadIdx.y;
#pragma unroll
  for (int j = 0; j < 4; ++j)
    t[ty + j * 8][tx] = src[(size_t)(by + ty + j * 8) * N + bx + tx];
  __syncthreads();
#pragma unroll
  for (int j = 0; j < 4; ++j)
    dst[(size_t)(bx + ty + j * 8) * K + by + tx] = (__bf16)t[tx][ty + j * 8];
}

// ---------------------------------------------------------------------------
// Tiled GEMM: C = A(MxK) * Bt^T, A bf16 row-major, Bt bf16 [N][K] row-major.
// OMODE: 0 = bf16 [b,h,s,d] | 1 = bf16 [b,h,d,s] | 2 = f32 row-major [M][N]
// Block: 256 threads (8 waves). Tile 128x128, K-step 32, double-buffered LDS
// filled with async global->LDS b128 copies. Wave tile: 32(M) x 64(N),
// i.e. 2 A-frags x 4 B-frags -> 8 WMMAs per wave per K-step.
// ---------------------------------------------------------------------------
template <int OMODE>
__global__ __launch_bounds__(256) void gemm_bf16_wmma(
    const __bf16* __restrict__ A, const __bf16* __restrict__ Bt,
    void* __restrict__ Cptr, int M, int N, int K, int heads, int seq) {
  __shared__ __bf16 lA[2][128][48];  // [buf][m][k], row stride 96B (16B align)
  __shared__ __bf16 lB[2][128][48];  // [buf][n][k]

  const int tid   = threadIdx.x;
  const int lane  = tid & 31;
  const int wave  = tid >> 5;
  const int tileN = blockIdx.x * 128;
  const int tileM = blockIdx.y * 128;
  const int wm    = (wave >> 1) * 32;  // wave M offset: 0/32/64/96
  const int wn    = (wave & 1) * 64;   // wave N offset: 0/64

  const int lm  = lane & 15;
  const int k0  = (lane < 16) ? 0 : 8;   // unused; A-frag loads also use kk0
  const int kk0 = (lane < 16) ? 0 : 16;  // B-frag K base per lane half

  v8f acc[2][4];
#pragma unroll
  for (int mi = 0; mi < 2; ++mi)
#pragma unroll
    for (int nc = 0; nc < 4; ++nc)
#pragma unroll
      for (int i = 0; i < 8; ++i) acc[mi][nc][i] = 0.0f;

  // stage one 128x32 A tile + 128x32 Bt tile into LDS buffer `buf`
  auto stage = [&](int kb, int buf) {
#pragma unroll
    for (int i = 0; i < 2; ++i) {
      int idx = tid + i * 256;  // 0..511
      int row = idx >> 2;       // 0..127
      int c8  = idx & 3;        // x8 bf16 (16B)
      copy_b128_to_lds(A + (size_t)(tileM + row) * K + kb + c8 * 8,
                       &lA[buf][row][c8 * 8]);
      copy_b128_to_lds(Bt + (size_t)(tileN + row) * K + kb + c8 * 8,
                       &lB[buf][row][c8 * 8]);
    }
  };

  stage(0, 0);
  const int nsteps = K >> 5;
  for (int s = 0; s < nsteps; ++s) {
    const int buf = s & 1;
    wait_async_copies();
    __syncthreads();
    if (s + 1 < nsteps) stage((s + 1) << 5, buf ^ 1);

    v16bf af[2];
#pragma unroll
    for (int mi = 0; mi < 2; ++mi) {
      const __bf16* ar = &lA[buf][wm + mi * 16 + lm][0];
#pragma unroll
      for (int t = 0; t < 8; ++t) {
        af[mi][t]     = ar[kk0 + t];
        af[mi][8 + t] = ar[kk0 + 8 + t];
      }
    }
#pragma unroll
    for (int nc = 0; nc < 4; ++nc) {
      v16bf bfr;
      const __bf16* br = &lB[buf][wn + nc * 16 + lm][kk0];
#pragma unroll
      for (int t = 0; t < 16; ++t) bfr[t] = br[t];
      acc[0][nc] = __builtin_amdgcn_wmma_f32_16x16x32_bf16(
          false, af[0], false, bfr, (short)0, acc[0][nc], false, false);
      acc[1][nc] = __builtin_amdgcn_wmma_f32_16x16x32_bf16(
          false, af[1], false, bfr, (short)0, acc[1][nc], false, false);
    }
  }

  // epilogue: VGPR r holds M = r + 8*(lane/16), N = lane%16
  const int mh = (lane >> 4) << 3;
#pragma unroll
  for (int mi = 0; mi < 2; ++mi) {
#pragma unroll
    for (int nc = 0; nc < 4; ++nc) {
#pragma unroll
      for (int r = 0; r < 8; ++r) {
        int   m   = tileM + wm + mi * 16 + r + mh;
        int   n   = tileN + wn + nc * 16 + lm;
        float val = acc[mi][nc][r];
        if (OMODE == 2) {
          ((float*)Cptr)[(size_t)m * N + n] = val;
        } else {
          int b = m / seq, ss = m % seq;
          int h = n >> 7, d = n & 127;
          size_t off;
          if (OMODE == 0)
            off = ((((size_t)b * heads + h) * seq + ss) << 7) + d;  // [b,h,s,d]
          else
            off = ((((size_t)b * heads + h) << 7) + d) * seq + ss;  // [b,h,d,s]
          ((__bf16*)Cptr)[off] = (__bf16)val;
        }
      }
    }
  }
}

// ---------------------------------------------------------------------------
// Flash attention: Q [b,h,s,d] bf16, K [b,kv,s,d] bf16, Vt [b,kv,d,s] bf16.
// Output: bf16 row-major [b*s][NHEAD*HD] (feeds the final GEMM as A).
// Block: 128 threads (4 waves); wave owns 16 query rows; kv tiles of 32.
// ---------------------------------------------------------------------------
__global__ __launch_bounds__(128) void attn_fwd_wmma(
    const __bf16* __restrict__ Q, const __bf16* __restrict__ Kd,
    const __bf16* __restrict__ Vt, __bf16* __restrict__ O, int seq) {
  __shared__ __bf16 lP[4][16][48];  // per-wave P transpose staging

  const int lane  = threadIdx.x & 31;
  const int wave  = threadIdx.x >> 5;
  const int h     = blockIdx.y;
  const int b     = blockIdx.z;
  const int kvh   = h >> 2;  // GQA group = 4
  const int qbase = blockIdx.x * 64 + wave * 16;

  const __bf16* q  = Q  + ((size_t)b * NHEAD + h)   * seq * HD;
  const __bf16* k  = Kd + ((size_t)b * NKV   + kvh) * seq * HD;
  const __bf16* vt = Vt + ((size_t)b * NKV   + kvh) * HD  * seq;

  const int lm  = lane & 15;
  const int k0  = (lane < 16) ? 0 : 8;
  const int kk0 = (lane < 16) ? 0 : 16;

  v16bf qf[4];
#pragma unroll
  for (int dc = 0; dc < 4; ++dc) {
    const __bf16* qr = q + (size_t)(qbase + lm) * HD + dc * 32;
#pragma unroll
    for (int t = 0; t < 8; ++t) {
      qf[dc][t]     = qr[kk0 + t];
      qf[dc][8 + t] = qr[kk0 + 8 + t];
    }
  }

  v8f o[8];
#pragma unroll
  for (int c = 0; c < 8; ++c)
#pragma unroll
    for (int i = 0; i < 8; ++i) o[c][i] = 0.0f;

  float mrow[8], lrow[8];
#pragma unroll
  for (int r = 0; r < 8; ++r) { mrow[r] = -3.0e38f; lrow[r] = 0.0f; }

  const float scale = 0.08838834764831845f;  // 1/sqrt(128)

  for (int kb = 0; kb < seq; kb += 32) {
    v8f s0, s1;
#pragma unroll
    for (int i = 0; i < 8; ++i) { s0[i] = 0.0f; s1[i] = 0.0f; }
#pragma unroll
    for (int dc = 0; dc < 4; ++dc) {
      v16bf bk0, bk1;
      const __bf16* kr0 = k + (size_t)(kb + lm)      * HD + dc * 32 + kk0;
      const __bf16* kr1 = k + (size_t)(kb + 16 + lm) * HD + dc * 32 + kk0;
#pragma unroll
      for (int t = 0; t < 16; ++t) { bk0[t] = kr0[t]; bk1[t] = kr1[t]; }
      s0 = __builtin_amdgcn_wmma_f32_16x16x32_bf16(
          false, qf[dc], false, bk0, (short)0, s0, false, false);
      s1 = __builtin_amdgcn_wmma_f32_16x16x32_bf16(
          false, qf[dc], false, bk1, (short)0, s1, false, false);
    }

    // online softmax; row r lives in lanes of the same half-wave
#pragma unroll
    for (int r = 0; r < 8; ++r) {
      float a0 = s0[r] * scale, a1 = s1[r] * scale;
      float rowmax = fmaxf(a0, a1);
#pragma unroll
      for (int mm = 1; mm < 16; mm <<= 1)
        rowmax = fmaxf(rowmax, __shfl_xor(rowmax, mm, 32));
      float mnew   = fmaxf(mrow[r], rowmax);
      float p0     = __expf(a0 - mnew);
      float p1     = __expf(a1 - mnew);
      float factor = __expf(mrow[r] - mnew);
      float rowsum = p0 + p1;
#pragma unroll
      for (int mm = 1; mm < 16; mm <<= 1)
        rowsum += __shfl_xor(rowsum, mm, 32);
      lrow[r] = lrow[r] * factor + rowsum;
      mrow[r] = mnew;
#pragma unroll
      for (int c = 0; c < 8; ++c) o[c][r] *= factor;
      s0[r] = p0;
      s1[r] = p1;
    }

    // transpose P (C layout -> A layout) via per-wave LDS (in-order per wave)
#pragma unroll
    for (int r = 0; r < 8; ++r) {
      int row = r + ((lane >> 4) << 3);
      lP[wave][row][lm]      = (__bf16)s0[r];
      lP[wave][row][16 + lm] = (__bf16)s1[r];
    }
    v16bf pf;
    {
      const __bf16* pr = &lP[wave][lm][0];
#pragma unroll
      for (int t = 0; t < 8; ++t) {
        pf[t]     = pr[kk0 + t];
        pf[8 + t] = pr[kk0 + 8 + t];
      }
    }

    // O += P * V ; B-fragment = contiguous rows of Vt
#pragma unroll
    for (int c = 0; c < 8; ++c) {
      v16bf bv;
      const __bf16* vr = vt + (size_t)(c * 16 + lm) * seq + kb + kk0;
#pragma unroll
      for (int t = 0; t < 16; ++t) bv[t] = vr[t];
      o[c] = __builtin_amdgcn_wmma_f32_16x16x32_bf16(
          false, pf, false, bv, (short)0, o[c], false, false);
    }
  }

#pragma unroll
  for (int c = 0; c < 8; ++c) {
#pragma unroll
    for (int r = 0; r < 8; ++r) {
      int   srow = qbase + r + ((lane >> 4) << 3);
      float val  = o[c][r] / lrow[r];
      O[((size_t)b * seq + srow) * HIDDEN + h * HD + c * 16 + lm] = (__bf16)val;
    }
  }
}

// ---------------------------------------------------------------------------
extern "C" void kernel_launch(void* const* d_in, const int* in_sizes, int n_in,
                              void* d_out, int out_size, void* d_ws,
                              size_t ws_size, hipStream_t stream) {
  const float* x   = (const float*)d_in[0];
  const float* w_q = (const float*)d_in[1];
  const float* w_k = (const float*)d_in[2];
  const float* w_v = (const float*)d_in[3];
  const float* w_o = (const float*)d_in[4];

  const int M   = in_sizes[0] / HIDDEN;  // bs * seq = 4096
  const int seq = 2048;
  const int bs  = M / seq;

  // bf16 workspace layout (~201 MB):
  // Xb | WqT | WkT | WvT | WoT | Q | K | Vt | attn-out
  __bf16* Xb  = (__bf16*)d_ws;
  __bf16* Wqt = Xb  + (size_t)M * HIDDEN;
  __bf16* Wkt = Wqt + (size_t)HIDDEN * HIDDEN;
  __bf16* Wvt = Wkt + (size_t)KV_DIM * HIDDEN;
  __bf16* Wot = Wvt + (size_t)KV_DIM * HIDDEN;
  __bf16* Qb  = Wot + (size_t)HIDDEN * HIDDEN;
  __bf16* Kb  = Qb  + (size_t)M * HIDDEN;
  __bf16* Vtb = Kb  + (size_t)M * KV_DIM;
  __bf16* Ob  = Vtb + (size_t)M * KV_DIM;

  // --- pre-pass: convert x, transpose+convert weights (bf16) ---
  {
    int n4 = (M * HIDDEN) / 4;
    convert_f32_bf16<<<dim3((n4 + 255) / 256), dim3(256), 0, stream>>>(x, Xb, n4);
    dim3 tb(32, 8);
    transpose_f32_bf16<<<dim3(HIDDEN / 32, HIDDEN / 32), tb, 0, stream>>>(
        w_q, Wqt, HIDDEN, HIDDEN);
    transpose_f32_bf16<<<dim3(KV_DIM / 32, HIDDEN / 32), tb, 0, stream>>>(
        w_k, Wkt, HIDDEN, KV_DIM);
    transpose_f32_bf16<<<dim3(KV_DIM / 32, HIDDEN / 32), tb, 0, stream>>>(
        w_v, Wvt, HIDDEN, KV_DIM);
    transpose_f32_bf16<<<dim3(HIDDEN / 32, HIDDEN / 32), tb, 0, stream>>>(
        w_o, Wot, HIDDEN, HIDDEN);
  }

  dim3 blk(256);
  // Q = x @ w_q -> [b,h,s,d]
  gemm_bf16_wmma<0><<<dim3(HIDDEN / 128, M / 128), blk, 0, stream>>>(
      Xb, Wqt, Qb, M, HIDDEN, HIDDEN, NHEAD, seq);
  // K = x @ w_k -> [b,kv,s,d]
  gemm_bf16_wmma<0><<<dim3(KV_DIM / 128, M / 128), blk, 0, stream>>>(
      Xb, Wkt, Kb, M, KV_DIM, HIDDEN, NKV, seq);
  // V = x @ w_v -> [b,kv,d,s] (pre-transposed for attention B-fragments)
  gemm_bf16_wmma<1><<<dim3(KV_DIM / 128, M / 128), blk, 0, stream>>>(
      Xb, Wvt, Vtb, M, KV_DIM, HIDDEN, NKV, seq);
  // flash attention -> bf16 [b*s][4096]
  attn_fwd_wmma<<<dim3(seq / 64, NHEAD, bs), dim3(128), 0, stream>>>(
      Qb, Kb, Vtb, Ob, seq);
  // out = attn @ w_o -> f32 [M][4096]
  gemm_bf16_wmma<2><<<dim3(HIDDEN / 128, M / 128), blk, 0, stream>>>(
      Ob, Wot, (float*)d_out, M, HIDDEN, HIDDEN, 0, seq);
}
